// LuminanceAwareMHSA_37125697306778
// MI455X (gfx1250) — matrix operand order repaired and checked by reading, not compiled, not run on hardware
//
#include <hip/hip_runtime.h>
#include <hip/hip_bf16.h>

typedef __attribute__((ext_vector_type(16))) _Float16 v16h;
typedef __attribute__((ext_vector_type(8)))  _Float16 v8h;
typedef __attribute__((ext_vector_type(8)))  float    v8f;

#define NB    4
#define NC    256
#define NH    48
#define NW    48
#define NPIX  2304
#define NHEAD 8
#define NDH   32
#define NIN   256
#define NHID  128
// SCALE * log2(e): softmax runs in the exp2 domain (v_exp_f32 is natively 2^x)
#define ATT_SCALE_LOG2 0.2550357078f
#define LOG2E 1.4426950408889634f
#define EPS_F 1e-6f
#define PSTRIDE 72   // P-tile LDS row stride in halfs (16B aligned, bank-conflict free)

__device__ __forceinline__ v8f wmma16(v16h a, v16h b, v8f c) {
  return __builtin_amdgcn_wmma_f32_16x16x32_f16(false, a, false, b, (short)0, c, false, false);
}

// ---------- Kernel 1: luma normalization + pooled inverse-luma bias ----------
// biasL is pre-multiplied by alpha*log2(e) so attention softmax can stay in exp2 domain.
__global__ __launch_bounds__(256)
void k_luma(const float* __restrict__ rgb, const float* __restrict__ alphap,
            float* __restrict__ luma_n, float* __restrict__ biasL) {
  __shared__ float sl[NPIX];
  __shared__ float sp[NPIX];
  __shared__ float red[256];
  const int b = blockIdx.x;
  const int tid = threadIdx.x;
  const float* rp = rgb + (size_t)b * 3 * NPIX;
  float lmin = 1e30f, lmax = -1e30f;
  for (int i = tid; i < NPIX; i += 256) {
    float y = 0.299f * rp[i] + 0.587f * rp[NPIX + i] + 0.114f * rp[2 * NPIX + i];
    sl[i] = y;
    lmin = fminf(lmin, y);
    lmax = fmaxf(lmax, y);
  }
  red[tid] = lmin; __syncthreads();
  for (int s = 128; s > 0; s >>= 1) { if (tid < s) red[tid] = fminf(red[tid], red[tid + s]); __syncthreads(); }
  lmin = red[0]; __syncthreads();
  red[tid] = lmax; __syncthreads();
  for (int s = 128; s > 0; s >>= 1) { if (tid < s) red[tid] = fmaxf(red[tid], red[tid + s]); __syncthreads(); }
  lmax = red[0]; __syncthreads();
  const float inv = 1.0f / (lmax - lmin + EPS_F);
  for (int i = tid; i < NPIX; i += 256) {
    float v = (sl[i] - lmin) * inv;
    sl[i] = v;
    luma_n[(size_t)b * NPIX + i] = v;
  }
  __syncthreads();
  float lsum = 0.f;
  for (int i = tid; i < NPIX; i += 256) {
    int py = i / NW, px = i % NW;
    float acc = 0.f;
    #pragma unroll
    for (int dy = -1; dy <= 1; ++dy)
      #pragma unroll
      for (int dx = -1; dx <= 1; ++dx) {
        int yy = py + dy, xx = px + dx;
        if (yy >= 0 && yy < NH && xx >= 0 && xx < NW) acc += 1.0f - sl[yy * NW + xx];
      }
    acc *= (1.0f / 9.0f);
    sp[i] = acc;
    lsum += acc;
  }
  red[tid] = lsum; __syncthreads();
  for (int s = 128; s > 0; s >>= 1) { if (tid < s) red[tid] += red[tid + s]; __syncthreads(); }
  const float mean = red[0] * (1.0f / NPIX);
  const float alpha2 = (*alphap) * LOG2E;
  for (int i = tid; i < NPIX; i += 256)
    biasL[(size_t)b * NPIX + i] = alpha2 * (sp[i] - mean);
}

// ---------- Kernel 2: conv1 3x3 (1 -> HID) + ReLU ----------
__global__ __launch_bounds__(256)
void k_conv1(const float* __restrict__ luma_n, const float* __restrict__ w,
             const float* __restrict__ bias, float* __restrict__ h1) {
  __shared__ float sl[NPIX];
  const int co = blockIdx.x, b = blockIdx.y, tid = threadIdx.x;
  for (int i = tid; i < NPIX; i += 256) sl[i] = luma_n[(size_t)b * NPIX + i];
  __syncthreads();
  float wk[9];
  #pragma unroll
  for (int t = 0; t < 9; ++t) wk[t] = w[co * 9 + t];
  const float bb = bias[co];
  float* out = h1 + ((size_t)b * NHID + co) * NPIX;
  for (int i = tid; i < NPIX; i += 256) {
    int py = i / NW, px = i % NW;
    float acc = bb;
    #pragma unroll
    for (int ky = 0; ky < 3; ++ky)
      #pragma unroll
      for (int kx = 0; kx < 3; ++kx) {
        int yy = py + ky - 1, xx = px + kx - 1;
        if (yy >= 0 && yy < NH && xx >= 0 && xx < NW) acc += wk[ky * 3 + kx] * sl[yy * NW + xx];
      }
    out[i] = fmaxf(acc, 0.f);
  }
}

// ---------- Kernel 3: conv2 3x3 (HID -> HID) + ReLU + spatial mean ----------
__global__ __launch_bounds__(256)
void k_conv2(const float* __restrict__ h1, const float* __restrict__ w,
             const float* __restrict__ bias, float* __restrict__ hm) {
  __shared__ float sp[NPIX];
  __shared__ float red[256];
  const int co = blockIdx.x, b = blockIdx.y, tid = threadIdx.x;
  float acc[9];
  #pragma unroll
  for (int p = 0; p < 9; ++p) acc[p] = 0.f;
  int py[9], px[9];
  #pragma unroll
  for (int p = 0; p < 9; ++p) { int i = p * 256 + tid; py[p] = i / NW; px[p] = i % NW; }
  for (int ci = 0; ci < NHID; ++ci) {
    __syncthreads();
    const float* plane = h1 + ((size_t)b * NHID + ci) * NPIX;
    for (int i = tid; i < NPIX; i += 256) sp[i] = plane[i];
    __syncthreads();
    float wl[9];
    #pragma unroll
    for (int t = 0; t < 9; ++t) wl[t] = w[((size_t)co * NHID + ci) * 9 + t];
    #pragma unroll
    for (int p = 0; p < 9; ++p) {
      #pragma unroll
      for (int ky = 0; ky < 3; ++ky) {
        int yy = py[p] + ky - 1;
        if (yy < 0 || yy >= NH) continue;
        #pragma unroll
        for (int kx = 0; kx < 3; ++kx) {
          int xx = px[p] + kx - 1;
          if (xx >= 0 && xx < NW) acc[p] += wl[ky * 3 + kx] * sp[yy * NW + xx];
        }
      }
    }
  }
  const float bb = bias[co];
  float s = 0.f;
  #pragma unroll
  for (int p = 0; p < 9; ++p) s += fmaxf(acc[p] + bb, 0.f);
  red[tid] = s; __syncthreads();
  for (int st = 128; st > 0; st >>= 1) { if (tid < st) red[tid] += red[tid + st]; __syncthreads(); }
  if (tid == 0) hm[(size_t)b * NHID + co] = red[0] * (1.0f / NPIX);
}

// ---------- Kernel 4: FiLM parameters (6 tiny GEMVs) ----------
__global__ __launch_bounds__(256)
void k_film(const float* __restrict__ hm,
            const float* __restrict__ gqw, const float* __restrict__ gqb,
            const float* __restrict__ bqw, const float* __restrict__ bqb,
            const float* __restrict__ gkw, const float* __restrict__ gkb,
            const float* __restrict__ bkw, const float* __restrict__ bkb,
            const float* __restrict__ gvw, const float* __restrict__ gvb,
            const float* __restrict__ bvw, const float* __restrict__ bvb,
            float* __restrict__ film) {
  __shared__ float sh[NHID];
  const int b = blockIdx.x, o = threadIdx.x;
  if (o < NHID) sh[o] = hm[(size_t)b * NHID + o];
  __syncthreads();
  float a0 = 0, a1 = 0, a2 = 0, a3 = 0, a4 = 0, a5 = 0;
  for (int i = 0; i < NHID; ++i) {
    float h = sh[i];
    a0 += gqw[o * NHID + i] * h; a1 += bqw[o * NHID + i] * h;
    a2 += gkw[o * NHID + i] * h; a3 += bkw[o * NHID + i] * h;
    a4 += gvw[o * NHID + i] * h; a5 += bvw[o * NHID + i] * h;
  }
  film[(0 * NB + b) * NIN + o] = a0 + gqb[o];
  film[(1 * NB + b) * NIN + o] = a1 + bqb[o];
  film[(2 * NB + b) * NIN + o] = a2 + gkb[o];
  film[(3 * NB + b) * NIN + o] = a3 + bkb[o];
  film[(4 * NB + b) * NIN + o] = a4 + gvb[o];
  film[(5 * NB + b) * NIN + o] = a5 + bvb[o];
}

// ---------- Kernel 5: QKV projection GEMM (WMMA) + FiLM, f16 pack ----------
// q,k layout: [B, HEADS, N, DH] f16 ; v layout: [B, HEADS, DH, N] f16
__global__ __launch_bounds__(256)
void k_qkv(const float* __restrict__ x,
           const float* __restrict__ wq, const float* __restrict__ bq,
           const float* __restrict__ wk, const float* __restrict__ bk,
           const float* __restrict__ wvp, const float* __restrict__ bv,
           const float* __restrict__ film,
           _Float16* __restrict__ qb, _Float16* __restrict__ kb,
           _Float16* __restrict__ vb) {
  __shared__ _Float16 sx[16 * 40];   // [n][c] tile, padded stride 40 halfs
  const int lane = threadIdx.x & 31, wid = threadIdx.x >> 5;
  const int n0 = blockIdx.x * 16;
  const int otile = blockIdx.y * 128;
  const int mi = blockIdx.z % 3, b = blockIdx.z / 3;
  const float* W; const float* bias;
  if (mi == 0)      { W = wq;  bias = bq; }
  else if (mi == 1) { W = wk;  bias = bk; }
  else              { W = wvp; bias = bv; }
  const float* gamma = film + (size_t)(2 * mi * NB + b) * NIN;
  const float* beta  = film + (size_t)((2 * mi + 1) * NB + b) * NIN;
  const float* xb = x + (size_t)b * NC * NPIX;
  const int m = lane & 15, koff8 = (lane >> 4) * 8, koff16 = (lane >> 4) * 16;
  const int o_wave = otile + wid * 16;
  // static coalesced staging map: element e -> (nn = e&15, cc = e>>4)
  const int nnA = threadIdx.x & 15, ccA = threadIdx.x >> 4;          // cc 0..15
  const int nnB = nnA,              ccB = ccA + 16;                  // cc 16..31
  v8f acc = {};
  for (int c0 = 0; c0 < NC; c0 += 32) {
    __syncthreads();
    sx[nnA * 40 + ccA] = (_Float16)xb[(size_t)(c0 + ccA) * NPIX + n0 + nnA];
    sx[nnB * 40 + ccB] = (_Float16)xb[(size_t)(c0 + ccB) * NPIX + n0 + nnB];
    __syncthreads();
    v16h a;
    const float* wr = W + (size_t)(o_wave + m) * NC + c0;
    #pragma unroll
    for (int j = 0; j < 8; ++j) {
      a[j]     = (_Float16)wr[koff8 + j];
      a[j + 8] = (_Float16)wr[koff8 + 16 + j];
    }
    v16h bf;
    const _Float16* bp = &sx[m * 40 + koff8];
    *(v8h*)&bf       = *(const v8h*)bp;
    *((v8h*)&bf + 1) = *(const v8h*)(bp + 16);
    acc = wmma16(a, bf, acc);
  }
  const int nloc = lane & 15, moff = (lane >> 4) * 8;
  const int n = n0 + nloc;
  #pragma unroll
  for (int r = 0; r < 8; ++r) {
    int o = o_wave + moff + r;
    float val = gamma[o] * (acc[r] + bias[o]) + beta[o];
    int head = o >> 5, dh = o & 31;
    if (mi < 2) {
      _Float16* dst = (mi == 0) ? qb : kb;
      dst[(((size_t)b * NHEAD + head) * NPIX + n) * NDH + dh] = (_Float16)val;
    } else {
      vb[(((size_t)b * NHEAD + head) * NDH + dh) * NPIX + n] = (_Float16)val;
    }
  }
}

// ---------- Kernel 6: flash attention (WMMA, exp2-domain online softmax) ----
// 64-key blocks: 4 S-WMMAs + 4 PV-WMMAs per block; one row-reduction pass
// amortized over 64 keys.
__global__ __launch_bounds__(256)
void k_attn(const _Float16* __restrict__ qbuf, const _Float16* __restrict__ kbuf,
            const _Float16* __restrict__ vbuf, const float* __restrict__ biasL,
            _Float16* __restrict__ obuf) {
  __shared__ _Float16 sP[8][16 * PSTRIDE];  // per-wave P staging (D -> A layout)
  const int lane = threadIdx.x & 31, wid = threadIdx.x >> 5;
  const int h = blockIdx.y, b = blockIdx.z;
  const int q0 = blockIdx.x * 128 + wid * 16;
  const _Float16* Q = qbuf + ((size_t)b * NHEAD + h) * NPIX * NDH;
  const _Float16* K = kbuf + ((size_t)b * NHEAD + h) * NPIX * NDH;
  const _Float16* V = vbuf + ((size_t)b * NHEAD + h) * NDH * NPIX;
  const float* bL = biasL + (size_t)b * NPIX;
  const int nloc = lane & 15;
  const int koff8 = (lane >> 4) * 8, koff16 = (lane >> 4) * 16, moff = (lane >> 4) * 8;
  v16h qa;
  {
    const _Float16* qr = Q + (size_t)(q0 + nloc) * NDH;
    *(v8h*)&qa       = *(const v8h*)(qr + koff8);
    *((v8h*)&qa + 1) = *(const v8h*)(qr + koff8 + 16);
  }
  float mrow[8], lrow[8];
  #pragma unroll
  for (int r = 0; r < 8; ++r) { mrow[r] = -1e30f; lrow[r] = 0.f; }
  v8f acc0 = {}, acc1 = {};
  _Float16* pw = &sP[wid][0];
  for (int k0 = 0; k0 < NPIX; k0 += 64) {
    // prefetch next K/V tiles (speculative; OOB dropped by HW)
    __builtin_prefetch(K + (size_t)(k0 + 64 + nloc) * NDH, 0, 1);
    __builtin_prefetch(V + (size_t)nloc * NPIX + k0 + 64, 0, 1);
    // ---- S = Q K^T for 64 keys (4 WMMAs) ----
    v8f s[4];
    #pragma unroll
    for (int t = 0; t < 4; ++t) {
      v16h kf;
      const _Float16* kr = K + (size_t)(k0 + t * 16 + nloc) * NDH + koff8;
      *(v8h*)&kf       = *(const v8h*)kr;
      *((v8h*)&kf + 1) = *(const v8h*)(kr + 16);
      v8f z = {};
      s[t] = wmma16(qa, kf, z);
    }
    float bias0 = bL[k0 + nloc];
    float bias1 = bL[k0 + 16 + nloc];
    float bias2 = bL[k0 + 32 + nloc];
    float bias3 = bL[k0 + 48 + nloc];
    // ---- online softmax (exp2 domain) ----
    #pragma unroll
    for (int r = 0; r < 8; ++r) {
      float v0 = s[0][r] * ATT_SCALE_LOG2 + bias0;
      float v1 = s[1][r] * ATT_SCALE_LOG2 + bias1;
      float v2 = s[2][r] * ATT_SCALE_LOG2 + bias2;
      float v3 = s[3][r] * ATT_SCALE_LOG2 + bias3;
      float mx = fmaxf(fmaxf(v0, v1), fmaxf(v2, v3));
      mx = fmaxf(mx, __shfl_xor(mx, 1, 32));
      mx = fmaxf(mx, __shfl_xor(mx, 2, 32));
      mx = fmaxf(mx, __shfl_xor(mx, 4, 32));
      mx = fmaxf(mx, __shfl_xor(mx, 8, 32));
      const float mnew = fmaxf(mrow[r], mx);
      const float corr = __builtin_amdgcn_exp2f(mrow[r] - mnew);
      mrow[r] = mnew;
      const float p0 = __builtin_amdgcn_exp2f(v0 - mnew);
      const float p1 = __builtin_amdgcn_exp2f(v1 - mnew);
      const float p2 = __builtin_amdgcn_exp2f(v2 - mnew);
      const float p3 = __builtin_amdgcn_exp2f(v3 - mnew);
      float ps = (p0 + p1) + (p2 + p3);
      ps += __shfl_xor(ps, 1, 32);
      ps += __shfl_xor(ps, 2, 32);
      ps += __shfl_xor(ps, 4, 32);
      ps += __shfl_xor(ps, 8, 32);
      lrow[r] = lrow[r] * corr + ps;
      acc0[r] *= corr;
      acc1[r] *= corr;
      _Float16* prow = pw + (moff + r) * PSTRIDE + nloc;
      prow[0]  = (_Float16)p0;
      prow[16] = (_Float16)p1;
      prow[32] = (_Float16)p2;
      prow[48] = (_Float16)p3;
    }
    asm volatile("s_wait_dscnt 0" ::: "memory");  // wave-private LDS RAW
    // ---- reload P as A-fragments (keys 0..31 and 32..63) ----
    v16h pa0, pa1;
    {
      const _Float16* pr = pw + nloc * PSTRIDE + koff16;
      *(v8h*)&pa0       = *(const v8h*)pr;
      *((v8h*)&pa0 + 1) = *(const v8h*)(pr + 8);
      *(v8h*)&pa1       = *(const v8h*)(pr + 32);
      *((v8h*)&pa1 + 1) = *(const v8h*)(pr + 40);
    }
    // ---- PV accumulate: acc(dh-half) += P(key-half) * V ----
    #pragma unroll
    for (int kk = 0; kk < 2; ++kk) {
      v16h vf0, vf1;
      const _Float16* vr0 = V + (size_t)nloc * NPIX + k0 + kk * 32 + koff16;
      const _Float16* vr1 = V + (size_t)(16 + nloc) * NPIX + k0 + kk * 32 + koff16;
      *(v8h*)&vf0       = *(const v8h*)vr0;
      *((v8h*)&vf0 + 1) = *(const v8h*)(vr0 + 8);
      *(v8h*)&vf1       = *(const v8h*)vr1;
      *((v8h*)&vf1 + 1) = *(const v8h*)(vr1 + 8);
      const v16h pa = kk ? pa1 : pa0;
      acc0 = wmma16(pa, vf0, acc0);
      acc1 = wmma16(pa, vf1, acc1);
    }
  }
  #pragma unroll
  for (int r = 0; r < 8; ++r) {
    const float inv = 1.0f / lrow[r];
    const int q = q0 + moff + r;
    _Float16* orow = obuf + ((size_t)b * NPIX + q) * NIN + h * NDH;
    orow[nloc]      = (_Float16)(acc0[r] * inv);
    orow[16 + nloc] = (_Float16)(acc1[r] * inv);
  }
}

// ---------- Kernel 7: output projection (WMMA) ----------
__global__ __launch_bounds__(256)
void k_proj(const _Float16* __restrict__ obuf, const float* __restrict__ wproj,
            const float* __restrict__ bproj, float* __restrict__ out) {
  const int lane = threadIdx.x & 31, wid = threadIdx.x >> 5;
  const int n0 = blockIdx.x * 16;
  const int c0 = blockIdx.y * 128 + wid * 16;
  const int b = blockIdx.z;
  const int m = lane & 15, nloc = lane & 15;
  const int koff8 = (lane >> 4) * 8, koff16 = (lane >> 4) * 16, moff = (lane >> 4) * 8;
  v8f acc = {};
  for (int o0 = 0; o0 < NIN; o0 += 32) {
    v16h a;
    const float* wr = wproj + (size_t)(c0 + m) * NIN + o0;
    #pragma unroll
    for (int j = 0; j < 8; ++j) {
      a[j]     = (_Float16)wr[koff8 + j];
      a[j + 8] = (_Float16)wr[koff8 + 16 + j];
    }
    v16h bf;
    const _Float16* br = obuf + ((size_t)b * NPIX + n0 + nloc) * NIN + o0 + koff8;
    *(v8h*)&bf       = *(const v8h*)br;
    *((v8h*)&bf + 1) = *(const v8h*)(br + 16);
    acc = wmma16(a, bf, acc);
  }
  #pragma unroll
  for (int r = 0; r < 8; ++r) {
    const int co = c0 + moff + r;
    out[((size_t)b * NC + co) * NPIX + n0 + nloc] = acc[r] + bproj[co];
  }
}

extern "C" void kernel_launch(void* const* d_in, const int* in_sizes, int n_in,
                              void* d_out, int out_size, void* d_ws, size_t ws_size,
                              hipStream_t stream) {
  const float* x      = (const float*)d_in[0];
  const float* rgb    = (const float*)d_in[1];
  const float* wq     = (const float*)d_in[2];
  const float* bq     = (const float*)d_in[3];
  const float* wk     = (const float*)d_in[4];
  const float* bk     = (const float*)d_in[5];
  const float* wv     = (const float*)d_in[6];
  const float* bv     = (const float*)d_in[7];
  const float* wproj  = (const float*)d_in[8];
  const float* bproj  = (const float*)d_in[9];
  const float* c1w    = (const float*)d_in[10];
  const float* c1b    = (const float*)d_in[11];
  const float* c2w    = (const float*)d_in[12];
  const float* c2b    = (const float*)d_in[13];
  const float* gqw    = (const float*)d_in[14];
  const float* gqb    = (const float*)d_in[15];
  const float* bqw    = (const float*)d_in[16];
  const float* bqb    = (const float*)d_in[17];
  const float* gkw    = (const float*)d_in[18];
  const float* gkb    = (const float*)d_in[19];
  const float* bkw    = (const float*)d_in[20];
  const float* bkb    = (const float*)d_in[21];
  const float* gvw    = (const float*)d_in[22];
  const float* gvb    = (const float*)d_in[23];
  const float* bvw    = (const float*)d_in[24];
  const float* bvb    = (const float*)d_in[25];
  const float* alphap = (const float*)d_in[26];
  (void)in_sizes; (void)n_in; (void)out_size; (void)ws_size;

  char* wsp = (char*)d_ws;
  size_t off = 0;
  auto take = [&](size_t bytes) {
    char* p = wsp + off;
    off += (bytes + 255) & ~(size_t)255;
    return p;
  };
  float* luma  = (float*)take(sizeof(float) * NB * NPIX);
  float* biasL = (float*)take(sizeof(float) * NB * NPIX);
  float* h1    = (float*)take(sizeof(float) * (size_t)NB * NHID * NPIX);
  float* hm    = (float*)take(sizeof(float) * NB * NHID);
  float* film  = (float*)take(sizeof(float) * 6 * NB * NIN);
  _Float16* qb = (_Float16*)take(sizeof(_Float16) * (size_t)NB * NHEAD * NPIX * NDH);
  _Float16* kb = (_Float16*)take(sizeof(_Float16) * (size_t)NB * NHEAD * NPIX * NDH);
  _Float16* vb = (_Float16*)take(sizeof(_Float16) * (size_t)NB * NHEAD * NPIX * NDH);
  _Float16* ob = (_Float16*)take(sizeof(_Float16) * (size_t)NB * NPIX * NIN);

  k_luma<<<dim3(NB), 256, 0, stream>>>(rgb, alphap, luma, biasL);
  k_conv1<<<dim3(NHID, NB), 256, 0, stream>>>(luma, c1w, c1b, h1);
  k_conv2<<<dim3(NHID, NB), 256, 0, stream>>>(h1, c2w, c2b, hm);
  k_film<<<dim3(NB), 256, 0, stream>>>(hm, gqw, gqb, bqw, bqb,
                                       gkw, gkb, bkw, bkb,
                                       gvw, gvb, bvw, bvb, film);
  k_qkv<<<dim3(NPIX / 16, 2, NB * 3), 256, 0, stream>>>(x, wq, bq, wk, bk, wv, bv,
                                                        film, qb, kb, vb);
  k_attn<<<dim3(NPIX / 128, NHEAD, NB), 256, 0, stream>>>(qb, kb, vb, biasL, ob);
  k_proj<<<dim3(NPIX / 16, 2, NB), 256, 0, stream>>>(ob, wproj, bproj, (float*)d_out);
}
